// Encoder_15040975470809
// MI455X (gfx1250) — matrix operand order repaired and checked by reading, not compiled, not run on hardware
//
#include <hip/hip_runtime.h>

// ---------------- Problem constants (match reference) ----------------
constexpr int U_   = 1200;           // GRU units
constexpr int UPAD = 1216;           // U padded to multiple of 32 (WMMA K)
constexpr int E_   = 620;            // word size
constexpr int EPAD = 640;            // E padded to multiple of 32
constexpr int B_   = 128;            // batch
constexpr int T_   = 64;             // seq len
constexpr int N3U  = 3 * U_;         // 3600 = 75 * 48 (exact WMMA N tiling)
constexpr float LN_EPS = 1e-3f;

// ---------------- WMMA types ----------------
typedef __attribute__((ext_vector_type(16))) __bf16     v16bf;
typedef __attribute__((ext_vector_type(8)))  float       v8f;
typedef __attribute__((ext_vector_type(4)))  unsigned int u32x4;
typedef int v4i __attribute__((vector_size(16)));         // matches builtin pointee

union FragB16 { u32x4 q[2]; v16bf v; };

#if __has_builtin(__builtin_amdgcn_global_load_async_to_lds_b128)
#define HAVE_ASYNC_LDS 1
#else
#define HAVE_ASYNC_LDS 0
#endif

// async copy of one 16B chunk: global -> LDS (per-lane addresses)
__device__ inline void async_copy_b128(const void* g, void* l) {
#if HAVE_ASYNC_LDS
    __builtin_amdgcn_global_load_async_to_lds_b128(
        (__attribute__((address_space(1))) v4i*)g,
        (__attribute__((address_space(3))) v4i*)l, 0, 0);
#else
    *(u32x4*)l = *(const u32x4*)g;     // global_load_b128 + ds_store_b128
#endif
}

template <int N>
__device__ inline void wait_asynccnt_le() {
#if HAVE_ASYNC_LDS
#if __has_builtin(__builtin_amdgcn_s_wait_asynccnt)
    __builtin_amdgcn_s_wait_asynccnt(N);
#else
    asm volatile("s_wait_asynccnt %0" :: "n"(N) : "memory");
#endif
#endif
}

__device__ inline unsigned short f2bf(float f) {
    union { float f; unsigned int u; } x; x.f = f;
    unsigned int r = (x.u + 0x7FFFu + ((x.u >> 16) & 1u)) >> 16;   // RNE
    return (unsigned short)r;
}

__device__ inline void load_frag_a(FragB16& f, const unsigned short* p) {
    f.q[0] = *(const u32x4*)p;           // K = koff .. koff+7
    f.q[1] = *(const u32x4*)(p + 8);     // K = koff+8 .. koff+15
}
__device__ inline v8f wmma_bf16(const FragB16& a, const FragB16& b, v8f c) {
    return __builtin_amdgcn_wmma_f32_16x16x32_bf16(false, a.v, false, b.v,
                                                   (short)0, c, false, false);
}

// =====================================================================
// Weight repack: fp32 row-major [K,N] -> bf16 WMMA-packed B tiles.
// Tile = 32(K) x 16(N). B striping used here (contiguous K, same order as A; not the ISA table order):
//   lanes 0-15  hold K=0..15  (V0..V7, 2 per VGPR), column = lane
//   lanes 16-31 hold K=16..31 (V0..V7, 2 per VGPR), column = lane-16
// Per-tile memory: chunk0 = lanes x V0..V3 (lane stride 16B),
//                  chunk1 = lanes x V4..V7. Tile = 1024 B.
// Tiles ordered [nt][kt] so the K loop streams contiguously.
// =====================================================================
__global__ void pack_b_bf16(const float* __restrict__ src,
                            unsigned short* __restrict__ dst,
                            int K, int Kpad, int N)
{
    int idx = blockIdx.x * blockDim.x + threadIdx.x;
    int total = Kpad * N;
    if (idx >= total) return;
    int k = idx / N, n = idx - k * N;
    float val = (k < K) ? src[(size_t)k * N + n] : 0.0f;

    int kt = k >> 5, kk = k & 31;
    int nt = n >> 4, nn = n & 15;
    int lane = nn + ((kk >> 4) << 4);      // kk<16 -> lanes 0-15, else 16-31
    int d    = (kk & 15) >> 1;             // VGPR index 0..7
    int half = kk & 1;                     // low/high 16 bits
    size_t KT = (size_t)(Kpad >> 5);
    size_t off = (((size_t)nt * KT + kt) * 1024)
               + ((size_t)(d >> 2) * 512) + (size_t)lane * 16
               + (size_t)(d & 3) * 4 + (size_t)half * 2;
    *(unsigned short*)((char*)dst + off) = f2bf(val);
}

// =====================================================================
// Embedding gather + tanh. Writes fp32 embedding output and bf16
// activation copies (forward order + time-reversed for backward GRU).
// =====================================================================
__global__ void gather_tanh(const int* __restrict__ sent,
                            const float* __restrict__ embed,
                            float* __restrict__ emb_out,            // [B,T,E] fp32
                            unsigned short* __restrict__ emb_fw,    // [B*T, EPAD] bf16
                            unsigned short* __restrict__ emb_bw)    // time-reversed
{
    int idx = blockIdx.x * blockDim.x + threadIdx.x;
    if (idx >= B_ * T_ * EPAD) return;
    int e  = idx % EPAD;
    int bt = idx / EPAD;
    int b  = bt / T_, t = bt - b * T_;
    float v = 0.0f;
    if (e < E_) {
        int w = sent[bt];
        v = tanhf(embed[(size_t)w * E_ + e]);
        emb_out[(size_t)bt * E_ + e] = v;
    }
    unsigned short bv = f2bf(v);
    emb_fw[(size_t)bt * EPAD + e] = bv;
    emb_bw[((size_t)b * T_ + (T_ - 1 - t)) * EPAD + e] = bv;
}

// =====================================================================
// GEMM: C[M,N] = A[M,Kpad](bf16,row-major) * Bpacked + bias[N]
// Block = 128 threads = 4 waves covering 64 rows x 48 cols of ONE
// n-group; wave w computes m-tile w. The three 1KB B tiles per k-step
// are SHARED by all waves: staged into a double-buffered 6KB LDS
// region with global_load_async_to_lds_b128 (waves 0-2 own one tile
// each -> exactly 2 full-EXEC async ops per wave per stage, so
// per-wave ASYNCcnt is deterministic and s_wait_asynccnt(2) drains
// exactly the previous stage). Wave id goes through readfirstlane so
// the producer branch is a scalar s_cbranch (no EXEC masking).
// A fragments are register-double-buffered.
// Requires: M%64==0, N%48==0, Kpad%32==0. EXEC all-ones.
// =====================================================================
__global__ __launch_bounds__(128) void gemm_bf16_wmma(
    const unsigned short* __restrict__ A,
    const unsigned short* __restrict__ Bp,
    const float* __restrict__ bias,
    float* __restrict__ C,
    int M, int N, int Kpad, int nt_store)
{
    __shared__ u32x4 smem[2 * 192];                // 2 stages x 3 tiles x 64 chunks
    const int KT   = Kpad >> 5;
    const int lane = threadIdx.x & 31;
    // wave id as an SGPR value -> producer branches become scalar branches
    const int w    = __builtin_amdgcn_readfirstlane(threadIdx.x >> 5);
    const int NG   = N / 48;
    const int ng   = blockIdx.x % NG;
    const int mb   = blockIdx.x / NG;
    const int m0   = mb * 64 + w * 16;
    const int n0   = ng * 48;

    // A fragment, contiguous K (same order as packed B):
    //   lane 0-15 (row=lane):   K = k0..k0+15
    //   lane 16-31 (row=l-16):  K = k0+16..k0+31
    const int row  = lane & 15;
    const int koff = (lane >> 4) << 4;             // 0 or 16
    const unsigned short* aptr = A + (size_t)(m0 + row) * Kpad + koff;

    const size_t panel = (size_t)KT * 1024;        // one 16-col B panel
    const char* Bbase  = (const char*)Bp + (size_t)(n0 >> 4) * panel;

    v8f acc0 = {0.f,0.f,0.f,0.f,0.f,0.f,0.f,0.f};
    v8f acc1 = acc0, acc2 = acc0;

    // ---- prologue: stage 0 of B into LDS buffer 0; A k-step 0 ----
    if (w < 3) {
        const char* g = Bbase + (size_t)w * panel + (size_t)lane * 16;
        u32x4* l = &smem[w * 64 + lane];
        async_copy_b128(g, l);
        async_copy_b128(g + 512, l + 32);
    }
    FragB16 aCur, aNxt;
    load_frag_a(aCur, aptr); aptr += 32;

    for (int kt = 0; kt < KT; ++kt) {
        const int buf = kt & 1;
        if (kt + 1 < KT) {
            // issue next B stage into the other LDS buffer
            if (w < 3) {
                const char* g = Bbase + (size_t)w * panel
                              + (size_t)(kt + 1) * 1024 + (size_t)lane * 16;
                u32x4* l = &smem[(buf ^ 1) * 192 + w * 64 + lane];
                async_copy_b128(g, l);
                async_copy_b128(g + 512, l + 32);
            }
            load_frag_a(aNxt, aptr); aptr += 32;
            if (w < 3) wait_asynccnt_le<2>();      // drain stage kt only
        } else {
            if (w < 3) wait_asynccnt_le<0>();
        }
        __syncthreads();                           // stage kt visible to all

        const u32x4* ls = &smem[buf * 192];
        FragB16 b0f, b1f, b2f;
        b0f.q[0] = ls[lane];            b0f.q[1] = ls[32 + lane];
        b1f.q[0] = ls[64 + lane];       b1f.q[1] = ls[96 + lane];
        b2f.q[0] = ls[128 + lane];      b2f.q[1] = ls[160 + lane];

        acc0 = wmma_bf16(aCur, b0f, acc0);
        acc1 = wmma_bf16(aCur, b1f, acc1);
        acc2 = wmma_bf16(aCur, b2f, acc2);

        __syncthreads();                           // done reading stage kt
        aCur = aNxt;
    }

    // D layout: lane 0-15: VGPR r = D[r, lane]; lane 16-31: D[8+r, lane-16]
    const int col  = lane & 15;
    const int mrow = (lane >> 4) << 3;
    const float bs0 = bias[n0 + col];
    const float bs1 = bias[n0 + 16 + col];
    const float bs2 = bias[n0 + 32 + col];
    float* cb = C + (size_t)(m0 + mrow) * N + n0 + col;
    if (nt_store) {
        // streaming output (x_proj, ~118MB/dir): keep it out of L2's way
#pragma unroll
        for (int r = 0; r < 8; ++r) {
            __builtin_nontemporal_store(acc0[r] + bs0, &cb[(size_t)r * N]);
            __builtin_nontemporal_store(acc1[r] + bs1, &cb[(size_t)r * N + 16]);
            __builtin_nontemporal_store(acc2[r] + bs2, &cb[(size_t)r * N + 32]);
        }
    } else {
#pragma unroll
        for (int r = 0; r < 8; ++r) {
            cb[(size_t)r * N]      = acc0[r] + bs0;
            cb[(size_t)r * N + 16] = acc1[r] + bs1;
            cb[(size_t)r * N + 32] = acc2[r] + bs2;
        }
    }
}

// =====================================================================
// h0 = 0  =>  LN(h0) = ln0_beta. Also zero bf16 K padding once.
// =====================================================================
__global__ void init_hln(const float* __restrict__ ln0_b,
                         float* __restrict__ hln_f32,          // [B,U]
                         unsigned short* __restrict__ hln_bf)  // [B,UPAD]
{
    int idx = blockIdx.x * blockDim.x + threadIdx.x;
    if (idx >= B_ * UPAD) return;
    int u = idx % UPAD, b = idx / UPAD;
    if (u < U_) {
        float v = ln0_b[u];
        hln_f32[(size_t)b * U_ + u] = v;
        hln_bf[(size_t)b * UPAD + u] = f2bf(v);
    } else {
        hln_bf[(size_t)b * UPAD + u] = 0;
    }
}

__device__ inline void block_reduce2(float* sa, float* sb, float& a, float& b) {
    int tid = threadIdx.x;
    sa[tid] = a; sb[tid] = b;
    __syncthreads();
    for (int s = 128; s > 0; s >>= 1) {
        if (tid < s) { sa[tid] += sa[tid + s]; sb[tid] += sb[tid + s]; }
        __syncthreads();
    }
    a = sa[0]; b = sb[0];
    __syncthreads();
}

// =====================================================================
// Fused GRU pointwise update + candidate-LN + next-step state-LN.
// One workgroup per batch row (256 threads, U=1200 -> <=5 elems/thread).
//   z = sig(xt0+i0); r = sig(xt1+i1); a = xt2 + r*i2
//   hh = tanh(LN3(a)); h' = z*LN0(h) + (1-z)*hh
// then writes LN0(h') (fp32 carry + bf16 for next GEMM A operand).
// x_proj rows are read exactly once -> non-temporal loads.
// =====================================================================
__global__ __launch_bounds__(256) void gru_pointwise(
    const float* __restrict__ xproj,      // [B*T, 3U]
    const float* __restrict__ inner,      // [B, 3U]
    const float* __restrict__ ln3g, const float* __restrict__ ln3b,
    const float* __restrict__ ln0g, const float* __restrict__ ln0b,
    float* __restrict__ hln_f32,          // [B,U] in/out (LN of state)
    unsigned short* __restrict__ hln_bf,  // [B,UPAD] out
    float* __restrict__ final_out,        // d_out + dir offset, row stride 2U
    int t, int is_last)
{
    __shared__ float sa[256], sb[256];
    const int b   = blockIdx.x;
    const int tid = threadIdx.x;
    const float* xt = xproj + ((size_t)b * T_ + t) * N3U;
    const float* in = inner + (size_t)b * N3U;
    float* hl = hln_f32 + (size_t)b * U_;

    float aL[5], zL[5], hlL[5];
    float s1 = 0.f, s2 = 0.f;
#pragma unroll
    for (int i = 0; i < 5; ++i) {
        int u = tid + i * 256;
        if (u < U_) {
            float x0 = __builtin_nontemporal_load(&xt[u]);
            float x1 = __builtin_nontemporal_load(&xt[U_ + u]);
            float x2 = __builtin_nontemporal_load(&xt[2 * U_ + u]);
            float zi = 1.f / (1.f + __expf(-(x0 + in[u])));
            float ri = 1.f / (1.f + __expf(-(x1 + in[U_ + u])));
            float ai = x2 + ri * in[2 * U_ + u];
            aL[i] = ai; zL[i] = zi; hlL[i] = hl[u];
            s1 += ai; s2 += ai * ai;
        }
    }
    block_reduce2(sa, sb, s1, s2);
    float m   = s1 * (1.f / U_);
    float inv = rsqrtf(s2 * (1.f / U_) - m * m + LN_EPS);

    float hn[5];
    float t1 = 0.f, t2 = 0.f;
#pragma unroll
    for (int i = 0; i < 5; ++i) {
        int u = tid + i * 256;
        if (u < U_) {
            float hh = tanhf(ln3g[u] * (aL[i] - m) * inv + ln3b[u]);
            float h  = zL[i] * hlL[i] + (1.f - zL[i]) * hh;
            hn[i] = h; t1 += h; t2 += h * h;
            if (is_last) final_out[(size_t)b * (2 * U_) + u] = h;
        }
    }
    block_reduce2(sa, sb, t1, t2);
    float m2   = t1 * (1.f / U_);
    float inv2 = rsqrtf(t2 * (1.f / U_) - m2 * m2 + LN_EPS);
#pragma unroll
    for (int i = 0; i < 5; ++i) {
        int u = tid + i * 256;
        if (u < U_) {
            float v = ln0g[u] * (hn[i] - m2) * inv2 + ln0b[u];
            hl[u] = v;
            hln_bf[(size_t)b * UPAD + u] = f2bf(v);
        }
    }
}

// =====================================================================
extern "C" void kernel_launch(void* const* d_in, const int* in_sizes, int n_in,
                              void* d_out, int out_size, void* d_ws, size_t ws_size,
                              hipStream_t stream) {
    (void)in_sizes; (void)n_in; (void)out_size; (void)ws_size;

    const int*   sent      = (const int*)  d_in[0];
    const float* embed     = (const float*)d_in[1];
    const float* fw_kernel = (const float*)d_in[2];
    const float* fw_rker   = (const float*)d_in[3];
    const float* fw_bias   = (const float*)d_in[4];
    const float* fw_ln0_g  = (const float*)d_in[5];
    const float* fw_ln0_b  = (const float*)d_in[6];
    const float* fw_ln3_g  = (const float*)d_in[7];
    const float* fw_ln3_b  = (const float*)d_in[8];
    const float* bw_kernel = (const float*)d_in[9];
    const float* bw_rker   = (const float*)d_in[10];
    const float* bw_bias   = (const float*)d_in[11];
    const float* bw_ln0_g  = (const float*)d_in[12];
    const float* bw_ln0_b  = (const float*)d_in[13];
    const float* bw_ln3_g  = (const float*)d_in[14];
    const float* bw_ln3_b  = (const float*)d_in[15];

    float* out_tv  = (float*)d_out;                       // [B, 2U]
    float* out_emb = out_tv + (size_t)B_ * 2 * U_;        // [B, T, E]

    // ---- workspace carve-up (256B aligned) ----
    char* ws = (char*)d_ws;
    auto carve = [&](size_t bytes) -> char* {
        char* p = ws; ws += (bytes + 255) & ~(size_t)255; return p;
    };
    const size_t KT_E = EPAD / 32, KT_U = UPAD / 32, NT = N3U / 16;
    unsigned short* pk_fw  = (unsigned short*)carve(KT_E * NT * 1024);      // packed fw_kernel
    unsigned short* pk_bw  = (unsigned short*)carve(KT_E * NT * 1024);
    unsigned short* pr_fw  = (unsigned short*)carve(KT_U * NT * 1024);      // packed fw_rkernel
    unsigned short* pr_bw  = (unsigned short*)carve(KT_U * NT * 1024);
    unsigned short* emb_fw = (unsigned short*)carve((size_t)B_ * T_ * EPAD * 2);
    unsigned short* emb_bw = (unsigned short*)carve((size_t)B_ * T_ * EPAD * 2);
    float* xp_fw    = (float*)carve((size_t)B_ * T_ * N3U * 4);
    float* xp_bw    = (float*)carve((size_t)B_ * T_ * N3U * 4);
    float* inner_fw = (float*)carve((size_t)B_ * N3U * 4);
    float* inner_bw = (float*)carve((size_t)B_ * N3U * 4);
    float* hlf_fw   = (float*)carve((size_t)B_ * U_ * 4);
    float* hlf_bw   = (float*)carve((size_t)B_ * U_ * 4);
    unsigned short* hlb_fw = (unsigned short*)carve((size_t)B_ * UPAD * 2);
    unsigned short* hlb_bw = (unsigned short*)carve((size_t)B_ * UPAD * 2);

    // ---- weight repack (fp32 -> packed bf16 WMMA tiles) ----
    {
        int ne = EPAD * N3U, nu = UPAD * N3U;
        pack_b_bf16<<<(ne + 255) / 256, 256, 0, stream>>>(fw_kernel, pk_fw, E_, EPAD, N3U);
        pack_b_bf16<<<(ne + 255) / 256, 256, 0, stream>>>(bw_kernel, pk_bw, E_, EPAD, N3U);
        pack_b_bf16<<<(nu + 255) / 256, 256, 0, stream>>>(fw_rker,  pr_fw, U_, UPAD, N3U);
        pack_b_bf16<<<(nu + 255) / 256, 256, 0, stream>>>(bw_rker,  pr_bw, U_, UPAD, N3U);
    }

    // ---- embedding gather + tanh ----
    {
        int n = B_ * T_ * EPAD;
        gather_tanh<<<(n + 255) / 256, 256, 0, stream>>>(sent, embed, out_emb, emb_fw, emb_bw);
    }

    // ---- time-parallel input projection (both directions) ----
    {
        int blocks = ((B_ * T_) / 64) * (N3U / 48);            // 128*75 = 9600
        gemm_bf16_wmma<<<blocks, 128, 0, stream>>>(emb_fw, pk_fw, fw_bias, xp_fw, B_ * T_, N3U, EPAD, 1);
        gemm_bf16_wmma<<<blocks, 128, 0, stream>>>(emb_bw, pk_bw, bw_bias, xp_bw, B_ * T_, N3U, EPAD, 1);
    }

    // ---- h0 = 0 => LN(h0) = beta ----
    {
        int n = B_ * UPAD;
        init_hln<<<(n + 255) / 256, 256, 0, stream>>>(fw_ln0_b, hlf_fw, hlb_fw);
        init_hln<<<(n + 255) / 256, 256, 0, stream>>>(bw_ln0_b, hlf_bw, hlb_bw);
    }

    // ---- recurrent scan ----
    const int gblocks = (B_ / 64) * (N3U / 48);                // 2*75 = 150
    for (int t = 0; t < T_; ++t) {
        int last = (t == T_ - 1) ? 1 : 0;
        gemm_bf16_wmma<<<gblocks, 128, 0, stream>>>(hlb_fw, pr_fw, fw_bias + N3U, inner_fw, B_, N3U, UPAD, 0);
        gemm_bf16_wmma<<<gblocks, 128, 0, stream>>>(hlb_bw, pr_bw, bw_bias + N3U, inner_bw, B_, N3U, UPAD, 0);
        gru_pointwise<<<B_, 256, 0, stream>>>(xp_fw, inner_fw, fw_ln3_g, fw_ln3_b, fw_ln0_g, fw_ln0_b,
                                              hlf_fw, hlb_fw, out_tv,       t, last);
        gru_pointwise<<<B_, 256, 0, stream>>>(xp_bw, inner_bw, bw_ln3_g, bw_ln3_b, bw_ln0_g, bw_ln0_b,
                                              hlf_bw, hlb_bw, out_tv + U_,  t, last);
    }
}
